// LSTMStack_34325378630247
// MI455X (gfx1250) — matrix operand order repaired and checked by reading, not compiled, not run on hardware
//
#include <hip/hip_runtime.h>
#include <hip/hip_bf16.h>
#include <stdint.h>

typedef __attribute__((ext_vector_type(16))) __bf16 v16bf;
typedef __attribute__((ext_vector_type(8)))  __bf16 v8bf;
typedef __attribute__((ext_vector_type(8)))  float  v8f;

#define B_ 32
#define T_ 256
#define H_ 1024
#define L_ 2
#define FOURH_ 4096

__device__ __forceinline__ unsigned short f2bf(float f) {
  unsigned u = __float_as_uint(f);
  u += 0x7FFFu + ((u >> 16) & 1u);   // round-to-nearest-even
  return (unsigned short)(u >> 16);
}

__device__ __forceinline__ float sigmoid_f(float x) {
  return 1.0f / (1.0f + __expf(-x));
}
__device__ __forceinline__ float tanh_f(float x) {
  float e = __expf(-2.0f * fabsf(x));
  float t = (1.0f - e) / (1.0f + e);
  return copysignf(t, x);
}

// ---------------------------------------------------------------------------
// Pack fused [Wx; Wh] (K = 2048, N = 4096) into per-lane WMMA B-fragment order:
// pw[ (((l*4+g)*64 + cg)*64 + kb)*512 + lane*16 + e ] =
//     W[l][ K = kb*32 + (lane<16?0:8) + (e<8 ? e : e+8) ][ n = g*1024 + cg*16 + lane%16 ]
// ---------------------------------------------------------------------------
__global__ void pack_w(const float* __restrict__ Wx, const float* __restrict__ Wh,
                       unsigned short* __restrict__ pw) {
  unsigned tid  = blockIdx.x * blockDim.x + threadIdx.x;   // < 2^20
  unsigned lane = tid & 31u;
  unsigned kb   = (tid >> 5) & 63u;
  unsigned cg   = (tid >> 11) & 63u;
  unsigned g    = (tid >> 17) & 3u;
  unsigned l    = tid >> 19;
  unsigned n    = g * 1024u + cg * 16u + (lane & 15u);
  unsigned kbase = kb * 32u + ((lane & 16u) ? 8u : 0u);
  unsigned short* dst = pw + (size_t)tid * 16u;
#pragma unroll
  for (int e = 0; e < 16; ++e) {
    unsigned K = kbase + (unsigned)(e < 8 ? e : e + 8);
    float v = (K < 1024u)
                ? Wx[((size_t)l * 1024u + K) * FOURH_ + n]
                : Wh[((size_t)l * 1024u + (K - 1024u)) * FOURH_ + n];
    dst[e] = f2bf(v);
  }
}

__global__ void cvt_x(const float* __restrict__ x, unsigned short* __restrict__ xb,
                      unsigned n) {
  unsigned i = blockIdx.x * blockDim.x + threadIdx.x;
  if (i < n) xb[i] = f2bf(x[i]);
}

__global__ void zero_state(unsigned short* __restrict__ hstate,
                           float* __restrict__ cstate,
                           unsigned* __restrict__ ctr) {
  unsigned i = blockIdx.x * blockDim.x + threadIdx.x;
  if (i < (unsigned)(L_ * 2 * B_ * H_)) hstate[i] = 0;        // 131072 bf16
  if (i < (unsigned)(L_ * B_ * H_))     cstate[i] = 0.0f;     // 65536 f32
  if (i == 0) *ctr = 0u;
}

// ---------------------------------------------------------------------------
// Persistent recurrent kernel: 128 blocks x 128 threads = 512 waves.
// gid bits: [0]=kc (K chunk: 0 = x-half K<1024, 1 = h-half K>=1024),
//           [1]=mTile, [7:2]=cg, [8]=layer.
// Layer 1 runs pipelined one timestep behind layer 0 (257 phases total).
// Each group of 2 waves (kc 0/1) reduces partial accumulators through LDS;
// the kc==0 wave applies bias + gates and updates c/h state.
// ---------------------------------------------------------------------------
__global__ __launch_bounds__(128) void lstm_persist(
    const unsigned short* __restrict__ xbf,   // [B][T][H] bf16
    const unsigned short* __restrict__ pw,    // packed weights
    const float* __restrict__ bias,           // [L][4H]
    unsigned short* __restrict__ hstate,      // [L][2][B][H] bf16 (double buffered)
    float* __restrict__ cstate,               // [L][B][H] f32
    float* __restrict__ yout,                 // [B][T][H] f32
    float* __restrict__ ctail,                // [L][B][H] f32
    float* __restrict__ htail,                // [L][B][H] f32
    unsigned* __restrict__ ctr)
{
  __shared__ float lds_red[2 * 4 * 8 * 32];   // 2 groups x 4 gates x 8 rows x 32 lanes

  const unsigned lane  = threadIdx.x & 31u;
  const unsigned wib   = threadIdx.x >> 5;          // wave in block: 0..3
  const unsigned gid   = blockIdx.x * 4u + wib;     // 0..511
  const unsigned kc    = gid & 1u;                  // K chunk
  const unsigned mTile = (gid >> 1) & 1u;
  const unsigned cg    = (gid >> 2) & 63u;
  const unsigned l     = gid >> 8;                  // layer
  const unsigned grp   = wib >> 1;                  // group within block: 0..1

  const unsigned ln    = lane & 15u;
  const unsigned hu    = cg * 16u + ln;             // hidden unit column
  const unsigned aRow  = mTile * 16u + ln;          // batch row for A loads
  const unsigned koff  = (lane & 16u) ? 8u : 0u;    // A-layout half-wave K offset

  // bias for this layer's 4 gates at column hu
  float bval[4];
#pragma unroll
  for (int g = 0; g < 4; ++g)
    bval[g] = bias[(size_t)l * FOURH_ + (size_t)g * 1024u + hu];

  // Weight fragment pointers: kb range for this wave is [kc*32, kc*32+32).
  const unsigned short* wp[4];
#pragma unroll
  for (int g = 0; g < 4; ++g)
    wp[g] = pw + ((size_t)((l * 4 + (unsigned)g) * 64 + cg) * 64 + kc * 32) * 512
               + lane * 16u;

  for (int p = 0; p <= T_; ++p) {
    const bool active = (l == 0) ? (p < T_) : (p >= 1);
    const int  t      = (l == 0) ? p : (p - 1);

    v8f zero8 = {0.f, 0.f, 0.f, 0.f, 0.f, 0.f, 0.f, 0.f};
    v8f acc[4] = {zero8, zero8, zero8, zero8};

    if (active) {
      // A source for this wave's K chunk (rows of [x_t | h_{t-1}], bf16)
      const unsigned short* asrc;
      if (kc == 0) {
        // x half: layer0 -> input sequence; layer1 -> h0 written this phase-1
        asrc = (l == 0)
                 ? (xbf + ((size_t)aRow * T_ + (size_t)t) * H_)
                 : (hstate + ((size_t)(0 * 2 + ((t + 1) & 1)) * B_ + aRow) * H_);
      } else {
        // h half: this layer's own recurrent state (read parity t&1)
        asrc = hstate + ((size_t)(l * 2 + (t & 1)) * B_ + aRow) * H_;
      }

#pragma unroll 4
      for (int kb = 0; kb < 32; ++kb) {
        unsigned k0 = (unsigned)kb * 32u + koff;
        v8bf lo = *(const v8bf*)((const __bf16*)asrc + k0);
        v8bf hi = *(const v8bf*)((const __bf16*)asrc + k0 + 16);
        v16bf a = __builtin_shufflevector(lo, hi, 0, 1, 2, 3, 4, 5, 6, 7,
                                          8, 9, 10, 11, 12, 13, 14, 15);
        v16bf b0 = *(const v16bf*)((const __bf16*)(wp[0] + (size_t)kb * 512));
        v16bf b1 = *(const v16bf*)((const __bf16*)(wp[1] + (size_t)kb * 512));
        v16bf b2 = *(const v16bf*)((const __bf16*)(wp[2] + (size_t)kb * 512));
        v16bf b3 = *(const v16bf*)((const __bf16*)(wp[3] + (size_t)kb * 512));
        acc[0] = __builtin_amdgcn_wmma_f32_16x16x32_bf16(false, a, false, b0, (short)0, acc[0], false, false);
        acc[1] = __builtin_amdgcn_wmma_f32_16x16x32_bf16(false, a, false, b1, (short)0, acc[1], false, false);
        acc[2] = __builtin_amdgcn_wmma_f32_16x16x32_bf16(false, a, false, b2, (short)0, acc[2], false, false);
        acc[3] = __builtin_amdgcn_wmma_f32_16x16x32_bf16(false, a, false, b3, (short)0, acc[3], false, false);
      }
    }

    // split-K reduction: kc==1 wave publishes partials to LDS
    if (active && kc == 1) {
#pragma unroll
      for (int g = 0; g < 4; ++g)
#pragma unroll
        for (int r = 0; r < 8; ++r)
          lds_red[(((size_t)grp * 4 + g) * 8 + r) * 32 + lane] = acc[g][r];
    }
    __syncthreads();

    if (active && kc == 0) {
#pragma unroll
      for (int g = 0; g < 4; ++g)
#pragma unroll
        for (int r = 0; r < 8; ++r)
          acc[g][r] += lds_red[(((size_t)grp * 4 + g) * 8 + r) * 32 + lane];

      // Gate math + state update. C/D layout: VGPR r -> M = r + (lane>=16 ? 8:0).
      unsigned short* hw = hstate + (size_t)(l * 2 + ((t + 1) & 1)) * B_ * H_;
#pragma unroll
      for (int r = 0; r < 8; ++r) {
        unsigned b = mTile * 16u + (unsigned)r + ((lane & 16u) ? 8u : 0u);
        float zi = acc[0][r] + bval[0];
        float zf = acc[1][r] + bval[1];
        float zg = acc[2][r] + bval[2];
        float zo = acc[3][r] + bval[3];
        float ig = sigmoid_f(zi);
        float fg = sigmoid_f(zf);
        float gg = tanh_f(zg);
        float og = sigmoid_f(zo);
        size_t cidx = ((size_t)l * B_ + b) * H_ + hu;
        float cn = fg * cstate[cidx] + ig * gg;
        float hn = og * tanh_f(cn);
        cstate[cidx] = cn;
        hw[(size_t)b * H_ + hu] = f2bf(hn);
        if (l == 1) yout[((size_t)b * T_ + (size_t)t) * H_ + hu] = hn;
        if (t == T_ - 1) {
          ctail[cidx] = cn;
          htail[cidx] = hn;
        }
      }
    }

    // ---- device-wide barrier (monotonic counter, reset by init kernel) ----
    __threadfence();
    __syncthreads();
    if (threadIdx.x == 0) {
      atomicAdd(ctr, 1u);
      unsigned target = (unsigned)(p + 1) * gridDim.x;
      while (__hip_atomic_load(ctr, __ATOMIC_ACQUIRE, __HIP_MEMORY_SCOPE_AGENT) < target) {
        __builtin_amdgcn_s_sleep(1);
      }
    }
    __syncthreads();
    __threadfence();
  }
}

// ---------------------------------------------------------------------------
extern "C" void kernel_launch(void* const* d_in, const int* in_sizes, int n_in,
                              void* d_out, int out_size, void* d_ws, size_t ws_size,
                              hipStream_t stream) {
  const float* x    = (const float*)d_in[0];  // (B,T,H)
  const float* Wx   = (const float*)d_in[1];  // (L,H,4H)
  const float* Wh   = (const float*)d_in[2];  // (L,H,4H)
  const float* bias = (const float*)d_in[3];  // (L,4H)

  char* ws = (char*)d_ws;
  // layout: packed weights | x bf16 | h state | c state | barrier counter
  unsigned short* pw     = (unsigned short*)(ws);                         // 33,554,432 B
  unsigned short* xbf    = (unsigned short*)(ws + 33554432u);             // 16,777,216 B
  unsigned short* hstate = (unsigned short*)(ws + 50331648u);             //    262,144 B
  float*          cstate = (float*)         (ws + 50331648u + 262144u);   //    262,144 B
  unsigned*       ctr    = (unsigned*)      (ws + 50331648u + 524288u);   //          4 B

  float* yout  = (float*)d_out;                                  // B*T*H
  float* ctail = (float*)d_out + (size_t)B_ * T_ * H_;           // L*B*H
  float* htail = ctail + (size_t)L_ * B_ * H_;                   // L*B*H

  pack_w<<<4096, 256, 0, stream>>>(Wx, Wh, pw);
  cvt_x<<<32768, 256, 0, stream>>>(x, xbf, (unsigned)(B_ * T_ * H_));
  zero_state<<<512, 256, 0, stream>>>(hstate, cstate, ctr);
  lstm_persist<<<128, 128, 0, stream>>>(xbf, pw, bias, hstate, cstate,
                                        yout, ctail, htail, ctr);
}
